// XConvDecoder_25529285607671
// MI455X (gfx1250) — matrix operand order repaired and checked by reading, not compiled, not run on hardware
//
#include <hip/hip_runtime.h>
#include <math.h>

typedef _Float16 v16h __attribute__((ext_vector_type(16)));
typedef float    v8f  __attribute__((ext_vector_type(8)));
typedef float    v4f  __attribute__((ext_vector_type(4)));

constexpr int NTOT = 32768;      // total points (16 graphs x 2048)
constexpr int NPER = 2048;
constexpr int DIM  = 32;
constexpr int KNN  = 5;
constexpr int NK   = NTOT * KNN; // 163840

__device__ __forceinline__ float eluf(float x) { return x > 0.f ? x : expm1f(x); }

// ---------------------------------------------------------------------------
// prep: f32 -> f16 copy of pos + squared norms
// ---------------------------------------------------------------------------
__global__ __launch_bounds__(256) void prep_kernel(const float* __restrict__ pos,
                                                   _Float16* __restrict__ pf,
                                                   float* __restrict__ sq) {
  const int n = blockIdx.x * 256 + threadIdx.x;
  if (n >= NTOT) return;
  float s = 0.f;
  #pragma unroll
  for (int d = 0; d < DIM; ++d) {
    const float v = pos[(size_t)n * DIM + d];
    s = fmaf(v, v, s);
    pf[(size_t)n * DIM + d] = (_Float16)v;
  }
  sq[n] = s;
}

// ---------------------------------------------------------------------------
// KNN: per-graph top-5 via WMMA Gram tiles.
// One wave owns 16 query rows; loops over 128 candidate tiles of 16.
// Tile transpose through wave-private LDS with wave-local s_wait_dscnt sync
// (no block barriers). Both lane halves scan 8 candidates of the same query
// row; the two sorted top-5 lists are merged once at the end via shfl_xor.
// ---------------------------------------------------------------------------
__global__ __launch_bounds__(128) void knn_kernel(const _Float16* __restrict__ pf,
                                                  const float* __restrict__ sq,
                                                  int* __restrict__ colOut) {
  __shared__ float tile[4][256];
  const int lane = threadIdx.x & 31;
  const int wv   = threadIdx.x >> 5;
  const int qt   = blockIdx.x * 4 + wv;          // 0..2047 query tiles
  const int b    = qt >> 7;                      // graph id
  const int qbase = b * NPER + (qt & 127) * 16;
  const int hsel = lane >> 4;
  const int lrow = lane & 15;

  // A fragment: rows = 16 queries, K = 32 dims (ISA 16-bit A 16x32 layout)
  v16h afrag;
  {
    const _Float16* src = pf + (size_t)(qbase + lrow) * DIM;
    #pragma unroll
    for (int i = 0; i < 8; ++i) {
      afrag[i]     = src[hsel * 8 + i];        // K = 8*hsel + i
      afrag[8 + i] = src[16 + hsel * 8 + i];   // K = 16 + 8*hsel + i
    }
  }
  float sqQ[8];
  #pragma unroll
  for (int r = 0; r < 8; ++r) sqQ[r] = sq[qbase + 8 * hsel + r]; // m = r + 8*hsel

  const int qrow = qbase + lrow;
  float bd[5]; int bi[5];
  #pragma unroll
  for (int j = 0; j < 5; ++j) { bd[j] = 3.0e38f; bi[j] = qrow; }

  for (int ct = 0; ct < NPER / 16; ++ct) {
    const int cbase = b * NPER + ct * 16;
    // B fragment: column n = lane&15 (candidate row), K in table order (same as A)
    v16h bfrag;
    {
      const _Float16* csrc = pf + (size_t)(cbase + lrow) * DIM;
      #pragma unroll
      for (int i = 0; i < 8; ++i) { bfrag[i] = csrc[hsel * 8 + i]; bfrag[8 + i] = csrc[16 + hsel * 8 + i]; }
    }
    if (ct + 1 < NPER / 16)  // next candidate tile -> global_prefetch_b8
      __builtin_prefetch(pf + (size_t)(cbase + 16 + lrow) * DIM, 0, 1);

    const float sqc = sq[cbase + lrow];
    v8f acc = {};
    acc = __builtin_amdgcn_wmma_f32_16x16x32_f16(false, afrag, false, bfrag,
                                                 (short)0, acc, false, false);
    #pragma unroll
    for (int r = 0; r < 8; ++r) {
      const float d = sqQ[r] + sqc - 2.0f * acc[r];
      tile[wv][(8 * hsel + r) * 16 + lrow] = d;   // [m][n]
    }
    // wave-private tile: DS ops are in-order within the wave; drain DScnt and
    // fence the compiler instead of a block barrier.
    asm volatile("s_wait_dscnt 0x0" ::: "memory");

    // every lane scans 8 candidates of query row lrow (cols hsel*8 .. +7)
    const v4f* rowp = (const v4f*)&tile[wv][lrow * 16 + hsel * 8];
    const v4f q0 = rowp[0];
    const v4f q1 = rowp[1];
    const float dv[8] = {q0[0], q0[1], q0[2], q0[3], q1[0], q1[1], q1[2], q1[3]};
    #pragma unroll
    for (int i = 0; i < 8; ++i) {
      float d = dv[i];
      const int gi = cbase + hsel * 8 + i;
      if (gi == qrow) d = -3.0e38f;               // self is always nearest
      if (d < bd[4]) {
        bd[4] = d; bi[4] = gi;
        #pragma unroll
        for (int t = 4; t > 0; --t) {
          if (bd[t] < bd[t - 1]) {
            const float td = bd[t]; bd[t] = bd[t - 1]; bd[t - 1] = td;
            const int   ti = bi[t]; bi[t] = bi[t - 1]; bi[t - 1] = ti;
          }
        }
      }
    }
    asm volatile("" ::: "memory");  // keep next iter's DS stores after reads
  }

  // merge the two per-half sorted top-5 lists (lane l <-> lane l+16)
  float od[5]; int oi[5];
  #pragma unroll
  for (int j = 0; j < 5; ++j) {
    od[j] = __shfl_xor(bd[j], 16, 32);
    oi[j] = __shfl_xor(bi[j], 16, 32);
  }
  if (lane < 16) {
    #pragma unroll
    for (int j = 0; j < 5; ++j) {
      const float d = od[j]; const int gi = oi[j];
      if (d < bd[4]) {
        bd[4] = d; bi[4] = gi;
        #pragma unroll
        for (int t = 4; t > 0; --t) {
          if (bd[t] < bd[t - 1]) {
            const float td = bd[t]; bd[t] = bd[t - 1]; bd[t - 1] = td;
            const int   ti = bi[t]; bi[t] = bi[t - 1]; bi[t - 1] = ti;
          }
        }
      }
    }
    #pragma unroll
    for (int j = 0; j < 5; ++j) colOut[(size_t)qrow * KNN + j] = bi[j];
  }
}

// ---------------------------------------------------------------------------
// deterministic per-block channel sum / sumsq partials (blockDim == 256)
// ---------------------------------------------------------------------------
template <int C>
__device__ __forceinline__ void write_partials(const float (&yv)[C],
                                               float* __restrict__ partials,
                                               float* __restrict__ red) {
  const int lane = threadIdx.x & 31;
  const int wv   = threadIdx.x >> 5;
  #pragma unroll
  for (int ch = 0; ch < C; ++ch) {
    float s = yv[ch];
    float q = yv[ch] * yv[ch];
    #pragma unroll
    for (int off = 16; off > 0; off >>= 1) {
      s += __shfl_xor(s, off, 32);
      q += __shfl_xor(q, off, 32);
    }
    if (lane == 0) { red[wv * 2 * C + ch] = s; red[wv * 2 * C + C + ch] = q; }
  }
  __syncthreads();
  if (threadIdx.x < C) {
    float S = 0.f, Q = 0.f;
    #pragma unroll
    for (int w = 0; w < 8; ++w) {
      S += red[w * 2 * C + threadIdx.x];
      Q += red[w * 2 * C + C + threadIdx.x];
    }
    partials[(size_t)blockIdx.x * 2 * C + threadIdx.x]     = S;
    partials[(size_t)blockIdx.x * 2 * C + C + threadIdx.x] = Q;
  }
  __syncthreads();
}

// partials -> BN scale/shift (scale = g*rsqrt(var+eps), shift = be - mean*scale)
__global__ void bn_finalize(const float* __restrict__ partials, int G, float invCnt,
                            const float* __restrict__ g, const float* __restrict__ be,
                            float* __restrict__ scale, float* __restrict__ shift, int C) {
  const int ch = threadIdx.x;
  if (ch >= C) return;
  float S = 0.f, Q = 0.f;
  for (int i = 0; i < G; ++i) {
    S += partials[(size_t)i * 2 * C + ch];
    Q += partials[(size_t)i * 2 * C + C + ch];
  }
  const float m  = S * invCnt;
  const float v  = Q * invCnt - m * m;
  const float sc = g[ch] * rsqrtf(v + 1e-5f);
  scale[ch] = sc;
  shift[ch] = be[ch] - m * sc;
}

// ---------------------------------------------------------------------------
// mlp1 stage 1: y = ELU(rel @ W1 + b1), rel computed from pos/col. rows = NK
// ---------------------------------------------------------------------------
template <int CD>
__global__ __launch_bounds__(256) void mlp1_stage1(const float* __restrict__ pos,
                                                   const int* __restrict__ col,
                                                   const float* __restrict__ W,
                                                   const float* __restrict__ b,
                                                   float* __restrict__ y,
                                                   float* __restrict__ partials) {
  __shared__ float Ws[DIM * CD];
  __shared__ float bs[CD];
  __shared__ float red[8 * 2 * CD];
  for (int i = threadIdx.x; i < DIM * CD; i += 256) Ws[i] = W[i];
  if (threadIdx.x < CD) bs[threadIdx.x] = b[threadIdx.x];
  __syncthreads();

  const int row = blockIdx.x * 256 + threadIdx.x;   // NK divisible by 256
  const int n = row / KNN;
  const int c = col[row];
  const float* pc = pos + (size_t)c * DIM;
  const float* pn = pos + (size_t)n * DIM;
  float rel[DIM];
  #pragma unroll
  for (int d = 0; d < DIM; ++d) rel[d] = pc[d] - pn[d];

  float yv[CD];
  #pragma unroll
  for (int ch = 0; ch < CD; ++ch) {
    float a = bs[ch];
    #pragma unroll
    for (int d = 0; d < DIM; ++d) a = fmaf(rel[d], Ws[d * CD + ch], a);
    yv[ch] = eluf(a);
    y[(size_t)row * CD + ch] = yv[ch];
  }
  write_partials<CD>(yv, partials, red);
}

// mlp1 stage 2: x = BN(y1); y2 = ELU(x @ W2 + b2). rows = NK
template <int CD>
__global__ __launch_bounds__(256) void mlp1_stage2(const float* __restrict__ y1,
                                                   const float* __restrict__ sc,
                                                   const float* __restrict__ sh,
                                                   const float* __restrict__ W,
                                                   const float* __restrict__ b,
                                                   float* __restrict__ y2,
                                                   float* __restrict__ partials) {
  __shared__ float Ws[CD * CD];
  __shared__ float bs[CD], scs[CD], shs[CD];
  __shared__ float red[8 * 2 * CD];
  for (int i = threadIdx.x; i < CD * CD; i += 256) Ws[i] = W[i];
  if (threadIdx.x < CD) {
    bs[threadIdx.x]  = b[threadIdx.x];
    scs[threadIdx.x] = sc[threadIdx.x];
    shs[threadIdx.x] = sh[threadIdx.x];
  }
  __syncthreads();

  const size_t row = (size_t)blockIdx.x * 256 + threadIdx.x;
  float xv[CD];
  #pragma unroll
  for (int ch = 0; ch < CD; ++ch) xv[ch] = y1[row * CD + ch] * scs[ch] + shs[ch];
  float yv[CD];
  #pragma unroll
  for (int ch = 0; ch < CD; ++ch) {
    float a = bs[ch];
    #pragma unroll
    for (int d = 0; d < CD; ++d) a = fmaf(xv[d], Ws[d * CD + ch], a);
    yv[ch] = eluf(a);
    y2[row * CD + ch] = yv[ch];
  }
  write_partials<CD>(yv, partials, red);
}

// mlp2 stage 1: t0 = ELU(rel.flatten(160) @ W(160x25) + b). rows = N
__global__ __launch_bounds__(256) void mlp2_stage1(const float* __restrict__ pos,
                                                   const int* __restrict__ col,
                                                   const float* __restrict__ W,
                                                   const float* __restrict__ b,
                                                   float* __restrict__ t0,
                                                   float* __restrict__ partials) {
  __shared__ float Ws[KNN * DIM * 25];   // 4000 floats
  __shared__ float bs[25];
  __shared__ float red[8 * 2 * 25];
  for (int i = threadIdx.x; i < KNN * DIM * 25; i += 256) Ws[i] = W[i];
  if (threadIdx.x < 25) bs[threadIdx.x] = b[threadIdx.x];
  __syncthreads();

  const int n = blockIdx.x * 256 + threadIdx.x;     // N divisible by 256
  float pn[DIM];
  #pragma unroll
  for (int d = 0; d < DIM; ++d) pn[d] = pos[(size_t)n * DIM + d];

  float acc[25];
  #pragma unroll
  for (int ch = 0; ch < 25; ++ch) acc[ch] = bs[ch];
  for (int k = 0; k < KNN; ++k) {
    const float* pc = pos + (size_t)col[(size_t)n * KNN + k] * DIM;
    for (int d = 0; d < DIM; ++d) {
      const float r = pc[d] - pn[d];
      const float* wrow = &Ws[(k * DIM + d) * 25];
      #pragma unroll
      for (int ch = 0; ch < 25; ++ch) acc[ch] = fmaf(r, wrow[ch], acc[ch]);
    }
  }
  float yv[25];
  #pragma unroll
  for (int ch = 0; ch < 25; ++ch) {
    yv[ch] = eluf(acc[ch]);
    t0[(size_t)n * 25 + ch] = yv[ch];
  }
  write_partials<25>(yv, partials, red);
}

// mlp2 grouped conv (full-width kernel): t1 = BN(tin); out[g][j] = sum_l t1[g][l]*cw[g][j][l] + cb[g][j]
__global__ __launch_bounds__(256) void mlp2_conv(const float* __restrict__ tin,
                                                 const float* __restrict__ sc,
                                                 const float* __restrict__ sh,
                                                 const float* __restrict__ cw,
                                                 const float* __restrict__ cb,
                                                 float* __restrict__ tout,
                                                 float* __restrict__ partials,
                                                 int doElu) {
  __shared__ float cws[125], cbs[25], scs[25], shs[25];
  __shared__ float red[8 * 2 * 25];
  if (threadIdx.x < 125) cws[threadIdx.x] = cw[threadIdx.x];
  if (threadIdx.x < 25) {
    cbs[threadIdx.x] = cb[threadIdx.x];
    scs[threadIdx.x] = sc[threadIdx.x];
    shs[threadIdx.x] = sh[threadIdx.x];
  }
  __syncthreads();

  const size_t n = (size_t)blockIdx.x * 256 + threadIdx.x;
  float t1[25];
  #pragma unroll
  for (int ch = 0; ch < 25; ++ch) t1[ch] = tin[n * 25 + ch] * scs[ch] + shs[ch];
  float yv[25];
  #pragma unroll
  for (int g = 0; g < 5; ++g) {
    #pragma unroll
    for (int j = 0; j < 5; ++j) {
      float a = cbs[g * 5 + j];
      #pragma unroll
      for (int l = 0; l < 5; ++l) a = fmaf(t1[g * 5 + l], cws[g * 25 + j * 5 + l], a);
      const float v = doElu ? eluf(a) : a;
      yv[g * 5 + j] = v;
      tout[n * 25 + g * 5 + j] = v;
    }
  }
  write_partials<25>(yv, partials, red);
}

// ---------------------------------------------------------------------------
// final: x_star = [BN(y2) | x[col]]  (C = CD+CIN channels x K)
//        xt = x_star @ T ; fc[c] = sum_j xt[c][j]*c_cw[c][j] + c_cb[c]
//        out = fc @ c_lw + c_lb  (optional ReLU).  rows = N
// ---------------------------------------------------------------------------
template <int CD, int CIN, int COUT, bool RELU>
__global__ __launch_bounds__(256) void final_kernel(const float* __restrict__ y2,
                                                    const float* __restrict__ sc2,
                                                    const float* __restrict__ sh2,
                                                    const float* __restrict__ xin,
                                                    const int* __restrict__ col,
                                                    const float* __restrict__ t4,
                                                    const float* __restrict__ scT,
                                                    const float* __restrict__ shT,
                                                    const float* __restrict__ ccw,
                                                    const float* __restrict__ ccb,
                                                    const float* __restrict__ clw,
                                                    const float* __restrict__ clb,
                                                    float* __restrict__ out) {
  constexpr int C = CD + CIN;
  __shared__ float cws[C * KNN];
  __shared__ float cbs[C];
  __shared__ float lws[C * COUT];
  __shared__ float lbs[COUT];
  __shared__ float s2[CD], h2[CD], sT[25], hT[25];
  for (int i = threadIdx.x; i < C * KNN; i += 256) cws[i] = ccw[i];
  for (int i = threadIdx.x; i < C * COUT; i += 256) lws[i] = clw[i];
  if (threadIdx.x < C) cbs[threadIdx.x] = ccb[threadIdx.x];
  if (threadIdx.x < COUT) lbs[threadIdx.x] = clb[threadIdx.x];
  if (threadIdx.x < CD) { s2[threadIdx.x] = sc2[threadIdx.x]; h2[threadIdx.x] = sh2[threadIdx.x]; }
  if (threadIdx.x < 25) { sT[threadIdx.x] = scT[threadIdx.x]; hT[threadIdx.x] = shT[threadIdx.x]; }
  __syncthreads();

  const size_t n = (size_t)blockIdx.x * 256 + threadIdx.x;
  float T[25];
  #pragma unroll
  for (int t = 0; t < 25; ++t) T[t] = t4[n * 25 + t] * sT[t] + hT[t];
  int cols[KNN];
  #pragma unroll
  for (int k = 0; k < KNN; ++k) cols[k] = col[n * KNN + k];

  float acc[COUT];
  #pragma unroll
  for (int o = 0; o < COUT; ++o) acc[o] = lbs[o];

  for (int c = 0; c < C; ++c) {
    float xs[KNN];
    if (c < CD) {
      #pragma unroll
      for (int k = 0; k < KNN; ++k)
        xs[k] = y2[(n * KNN + k) * CD + c] * s2[c] + h2[c];
    } else {
      #pragma unroll
      for (int k = 0; k < KNN; ++k)
        xs[k] = xin[(size_t)cols[k] * CIN + (c - CD)];
    }
    float fc = cbs[c];
    #pragma unroll
    for (int j = 0; j < KNN; ++j) {
      float xt = 0.f;
      #pragma unroll
      for (int k = 0; k < KNN; ++k) xt = fmaf(xs[k], T[k * 5 + j], xt);
      fc = fmaf(xt, cws[c * KNN + j], fc);
    }
    #pragma unroll
    for (int o = 0; o < COUT; ++o) acc[o] = fmaf(fc, lws[c * COUT + o], acc[o]);
  }
  #pragma unroll
  for (int o = 0; o < COUT; ++o)
    out[n * COUT + o] = RELU ? fmaxf(acc[o], 0.f) : acc[o];
}

// ---------------------------------------------------------------------------
// host-side orchestration of one XConv layer
// ---------------------------------------------------------------------------
template <int CD, int CIN, int COUT, bool RELU>
static void run_xconv(const float* const* P, const float* pos, const float* xin,
                      const int* col, float* y1, float* y2, float* tA, float* tB,
                      float* part, float* small, float* out, hipStream_t s) {
  const float *m1w1 = P[0],  *m1b1 = P[1],  *m1g1 = P[2],  *m1be1 = P[3];
  const float *m1w2 = P[4],  *m1b2 = P[5],  *m1g2 = P[6],  *m1be2 = P[7];
  const float *m2w1 = P[8],  *m2b1 = P[9],  *m2g1 = P[10], *m2be1 = P[11];
  const float *m2cw1 = P[12], *m2cb1 = P[13], *m2g2 = P[14], *m2be2 = P[15];
  const float *m2cw2 = P[16], *m2cb2 = P[17], *m2g3 = P[18], *m2be3 = P[19];
  const float *ccw = P[20], *ccb = P[21], *clw = P[22], *clb = P[23];

  float *sc1 = small,       *sh1 = small + 32;
  float *sc2 = small + 64,  *sh2 = small + 96;
  float *scA = small + 128, *shA = small + 160;
  float *scB = small + 192, *shB = small + 224;
  float *scT = small + 256, *shT = small + 288;

  const int GNK = NK / 256;     // 640
  const int GN  = NTOT / 256;   // 128

  mlp1_stage1<CD><<<GNK, 256, 0, s>>>(pos, col, m1w1, m1b1, y1, part);
  bn_finalize<<<1, CD, 0, s>>>(part, GNK, 1.f / NK, m1g1, m1be1, sc1, sh1, CD);
  mlp1_stage2<CD><<<GNK, 256, 0, s>>>(y1, sc1, sh1, m1w2, m1b2, y2, part);
  bn_finalize<<<1, CD, 0, s>>>(part, GNK, 1.f / NK, m1g2, m1be2, sc2, sh2, CD);

  mlp2_stage1<<<GN, 256, 0, s>>>(pos, col, m2w1, m2b1, tA, part);
  bn_finalize<<<1, 25, 0, s>>>(part, GN, 1.f / NTOT, m2g1, m2be1, scA, shA, 25);
  mlp2_conv<<<GN, 256, 0, s>>>(tA, scA, shA, m2cw1, m2cb1, tB, part, 1);
  bn_finalize<<<1, 25, 0, s>>>(part, GN, 1.f / NTOT, m2g2, m2be2, scB, shB, 25);
  mlp2_conv<<<GN, 256, 0, s>>>(tB, scB, shB, m2cw2, m2cb2, tA, part, 0);
  bn_finalize<<<1, 25, 0, s>>>(part, GN, 1.f / NTOT, m2g3, m2be3, scT, shT, 25);

  final_kernel<CD, CIN, COUT, RELU><<<GN, 256, 0, s>>>(
      y2, sc2, sh2, xin, col, tA, scT, shT, ccw, ccb, clw, clb, out);
}

extern "C" void kernel_launch(void* const* d_in, const int* in_sizes, int n_in,
                              void* d_out, int out_size, void* d_ws, size_t ws_size,
                              hipStream_t stream) {
  (void)in_sizes; (void)n_in; (void)out_size; (void)ws_size;

  const float* pos = (const float*)d_in[0];
  // d_in[1] = batch (contiguous equal-size graphs; implicit)
  const float* P1[24];
  const float* P2[24];
  for (int i = 0; i < 24; ++i) {
    P1[i] = (const float*)d_in[2 + i];
    P2[i] = (const float*)d_in[26 + i];
  }

  char* w = (char*)d_ws;
  size_t o = 0;
  auto take = [&](size_t bytes) -> void* {
    void* p = w + o;
    o += (bytes + 255) & ~(size_t)255;
    return p;
  };
  _Float16* pf  = (_Float16*)take((size_t)NTOT * DIM * sizeof(_Float16));
  float*    sq  = (float*)take((size_t)NTOT * sizeof(float));
  int*      col = (int*)take((size_t)NTOT * KNN * sizeof(int));
  float*    y1  = (float*)take((size_t)NK * 32 * sizeof(float));
  float*    y2  = (float*)take((size_t)NK * 32 * sizeof(float));
  float*    tA  = (float*)take((size_t)NTOT * 25 * sizeof(float));
  float*    tB  = (float*)take((size_t)NTOT * 25 * sizeof(float));
  float*    x1  = (float*)take((size_t)NTOT * 32 * sizeof(float));
  float*    part = (float*)take((size_t)(NK / 256) * 64 * sizeof(float));
  float*    small = (float*)take(4096);

  // 1) prep + KNN (WMMA)
  prep_kernel<<<NTOT / 256, 256, 0, stream>>>(pos, pf, sq);
  knn_kernel<<<(NTOT / 16) / 4, 128, 0, stream>>>(pf, sq, col);

  // 2) XConv layer 1: C_in=32 (x=pos), C_delta=32, C_out=32, ReLU
  run_xconv<32, 32, 32, true>(P1, pos, pos, col, y1, y2, tA, tB, part, small, x1, stream);

  // 3) XConv layer 2: C_in=32 (x=x1), C_delta=3, C_out=3
  run_xconv<3, 32, 3, false>(P2, pos, x1, col, y1, y2, tA, tB, part, small,
                             (float*)d_out, stream);
}
